// GeoGraphLyaerMPNN_12824772346489
// MI455X (gfx1250) — compile-verified, matrix operand order checked by reading, not run on hardware
//
#include <hip/hip_runtime.h>
#include <hip/hip_bf16.h>

typedef unsigned int   u32;
typedef unsigned short u16;

typedef __attribute__((ext_vector_type(16))) __bf16 v16bf;
typedef __attribute__((ext_vector_type(8)))  float  v8f;
typedef __attribute__((ext_vector_type(4)))  u32    v4u;

union Frag16 { v16bf v; v4u q[2]; };
union Acc8   { v8f v; float f[8]; };

__device__ __forceinline__ u16 f2bf(float f) {
  union { float f; u32 u; } c; c.f = f;
  u32 r = c.u + 0x7FFFu + ((c.u >> 16) & 1u);   // round-to-nearest-even
  return (u16)(r >> 16);
}
__device__ __forceinline__ float bf2f(u16 b) {
  union { u32 u; float f; } c; c.u = ((u32)b) << 16;
  return c.f;
}
__device__ __forceinline__ float sigmoidf_(float x) {
  return 1.0f / (1.0f + __expf(-x));
}
__device__ __forceinline__ void atomic_add_f32(float* p, float v) {
  __hip_atomic_fetch_add(p, v, __ATOMIC_RELAXED, __HIP_MEMORY_SCOPE_AGENT);
}

// ---------------------------------------------------------------------------
// Embedder: out[r, c] = relu(sum_k X[r,k]*W[k,c] + b[c]), bf16 output.
// K is tiny (1/16/64) -> streaming VALU kernel is bandwidth-optimal.
// ---------------------------------------------------------------------------
__global__ __launch_bounds__(128)
void embed_relu(int rows, int K, const float* __restrict__ X,
                const float* __restrict__ W, const float* __restrict__ B,
                u16* __restrict__ out) {
  __shared__ float sx[64];
  const int row = blockIdx.x;
  if (row >= rows) return;
  const int t = threadIdx.x;   // column 0..127
  if (t < K) sx[t] = X[(size_t)row * K + t];
  __syncthreads();
  float acc = B[t];
  for (int k = 0; k < K; ++k) acc = fmaf(sx[k], W[k * 128 + t], acc);
  out[(size_t)row * 128 + t] = f2bf(acc > 0.f ? acc : 0.f);
}

// ---------------------------------------------------------------------------
// Pack fp32 weights [K,128] into per-lane WMMA B-fragment layout (bf16).
// Fragment (nt, kb): lane L covers column n = nt*16 + L%16, half h = L/16,
// elems j<8 -> k = kb*32 + h*8+j ; j>=8 -> k = kb*32 + 16 + h*8 + (j-8).
// Stored as 32 contiguous bytes per lane -> B-frag = two 16B global loads.
// ---------------------------------------------------------------------------
__global__ __launch_bounds__(128)
void pack_w(const float* __restrict__ W, int KB, u16* __restrict__ out) {
  const int idx = blockIdx.x * blockDim.x + threadIdx.x;   // = (nt*KB+kb)*32+lane
  const int total = 8 * KB * 32;
  if (idx >= total) return;
  const int lane = idx & 31;
  const int kb   = (idx >> 5) % KB;
  const int nt   = idx / (32 * KB);
  const int n    = nt * 16 + (lane & 15);
  const int hf   = lane >> 4;
  u16* o = out + (size_t)idx * 16;
#pragma unroll
  for (int j = 0; j < 16; ++j) {
    const int k = kb * 32 + ((j < 8) ? (hf * 8 + j) : (16 + hf * 8 + (j - 8)));
    o[j] = f2bf(W[k * 128 + n]);
  }
}

__global__ void zero_f32(float* __restrict__ p, size_t n) {
  size_t i = (size_t)blockIdx.x * blockDim.x + threadIdx.x;
  if (i < n) p[i] = 0.f;
}

// ---------------------------------------------------------------------------
// Fused gather -> bf16 WMMA GEMM (K=256) -> bias -> scatter-atomic segment sum.
//   m[e,:] = [feat1[src[e]] ‖ feat2[e]] @ W + b ;  agg[dst[e],:] += m[e,:]
// Block: 128 threads = 4 waves, 64 rows; each wave owns a 16x128 C tile.
// ---------------------------------------------------------------------------
__global__ __launch_bounds__(128)
void msg_gemm_scatter(int rows,
                      const u16* __restrict__ feat1,
                      const u16* __restrict__ feat2,
                      const int* __restrict__ src,
                      const int* __restrict__ dst,
                      const u16* __restrict__ wpk,     // packed, KB=8
                      const float* __restrict__ bias,
                      float* __restrict__ agg) {
  __shared__ __align__(16) u16 sA[64 * 256];
  __shared__ int sSrc[64];
  __shared__ int sDst[64];
  const int t = threadIdx.x;
  const int base = blockIdx.x * 64;

  if (t < 64) {
    const int r = base + t;
    sSrc[t] = (r < rows) ? src[r] : 0;
    sDst[t] = (r < rows) ? dst[r] : 0;
  }
  __syncthreads();

  __builtin_prefetch(wpk, 0, 1);   // pull packed weights toward L2

  // stage A: 64 rows x 256 bf16 (128 dwords/row): [gathered feat1 | feat2]
  for (int i = t; i < 64 * 128; i += 128) {
    const int r = i >> 7, w = i & 127;
    const int grow = base + r;
    u32 val = 0;
    if (grow < rows) {
      if (w < 64) val = ((const u32*)(feat1 + (size_t)sSrc[r] * 128))[w];
      else        val = ((const u32*)(feat2 + (size_t)grow * 128))[w - 64];
    }
    ((u32*)sA)[i] = val;
  }
  __syncthreads();

  const int lane  = t & 31;
  const int wave  = t >> 5;
  const int mrow0 = wave * 16;
  const int mA    = lane & 15;
  const int hf    = lane >> 4;
  const u16* aRow = sA + (mrow0 + mA) * 256;

  Acc8 acc[8];
#pragma unroll
  for (int nt = 0; nt < 8; ++nt)
#pragma unroll
    for (int j = 0; j < 8; ++j) acc[nt].f[j] = 0.f;

  for (int kb = 0; kb < 8; ++kb) {
    Frag16 af;
    const u16* ap = aRow + kb * 32 + hf * 8;
    af.q[0] = *(const v4u*)(ap);
    af.q[1] = *(const v4u*)(ap + 16);
#pragma unroll
    for (int nt = 0; nt < 8; ++nt) {
      Frag16 bw;
      const v4u* bp = (const v4u*)wpk + ((size_t)((nt << 3) + kb) * 32 + lane) * 2;
      bw.q[0] = bp[0];
      bw.q[1] = bp[1];
      acc[nt].v = __builtin_amdgcn_wmma_f32_16x16x32_bf16(
          false, af.v, false, bw.v, (short)0, acc[nt].v, false, false);
    }
  }

  const int n = lane & 15;
#pragma unroll
  for (int nt = 0; nt < 8; ++nt) {
    const float bv = bias[nt * 16 + n];
#pragma unroll
    for (int v = 0; v < 8; ++v) {
      const int rloc = mrow0 + v + 8 * hf;
      const int grow = base + rloc;
      if (grow < rows) {
        atomic_add_f32(agg + (size_t)sDst[rloc] * 128 + nt * 16 + n,
                       acc[nt].f[v] + bv);
      }
    }
  }
}

// ---------------------------------------------------------------------------
// Fused update + gate + blend:
//   h    = relu([x ‖ agg] @ Wupd + bupd)            (K=256)
//   beta = sigmoid([x ‖ h ‖ x-h] @ Wgate + bgate)   (K=384, A built in LDS)
//   out  = beta*x + (1-beta)*h                      (bf16 output)
// ---------------------------------------------------------------------------
__global__ __launch_bounds__(128)
void upd_gate_blend(int rows,
                    const u16* __restrict__ xbuf,
                    const float* __restrict__ aggbuf,
                    const u16* __restrict__ wupd,    // packed KB=8
                    const float* __restrict__ bupd,
                    const u16* __restrict__ wgate,   // packed KB=12
                    const float* __restrict__ bgate,
                    u16* __restrict__ outbuf) {
  __shared__ __align__(16) u16 sA[64 * 384];   // [x | h | x-h]
  __shared__ __align__(16) u16 sG[64 * 128];   // agg as bf16
  const int t = threadIdx.x;
  const int base = blockIdx.x * 64;

  // stage x (cols 0..127 of sA)
  for (int i = t; i < 64 * 64; i += 128) {
    const int r = i >> 6, w = i & 63;
    const int grow = base + r;
    u32 val = 0;
    if (grow < rows) val = ((const u32*)(xbuf + (size_t)grow * 128))[w];
    ((u32*)sA)[r * 192 + w] = val;
  }
  // stage agg (fp32 -> bf16)
  for (int i = t; i < 64 * 128; i += 128) {
    const int r = i >> 7, c = i & 127;
    const int grow = base + r;
    const float g = (grow < rows) ? aggbuf[(size_t)grow * 128 + c] : 0.f;
    sG[i] = f2bf(g);
  }
  __syncthreads();

  const int lane  = t & 31;
  const int wave  = t >> 5;
  const int mrow0 = wave * 16;
  const int mA    = lane & 15;
  const int hf    = lane >> 4;
  const int rowIdx = mrow0 + mA;
  const int n = lane & 15;

  // ---- GEMM1: h = relu([x|agg] @ Wupd + bupd) ----
  Acc8 acc[8];
#pragma unroll
  for (int nt = 0; nt < 8; ++nt)
#pragma unroll
    for (int j = 0; j < 8; ++j) acc[nt].f[j] = 0.f;

  for (int kb = 0; kb < 8; ++kb) {
    const u16* ap = (kb < 4) ? (sA + rowIdx * 384 + kb * 32)
                             : (sG + rowIdx * 128 + (kb - 4) * 32);
    ap += hf * 8;
    Frag16 af;
    af.q[0] = *(const v4u*)(ap);
    af.q[1] = *(const v4u*)(ap + 16);
#pragma unroll
    for (int nt = 0; nt < 8; ++nt) {
      Frag16 bw;
      const v4u* bp = (const v4u*)wupd + ((size_t)((nt << 3) + kb) * 32 + lane) * 2;
      bw.q[0] = bp[0];
      bw.q[1] = bp[1];
      acc[nt].v = __builtin_amdgcn_wmma_f32_16x16x32_bf16(
          false, af.v, false, bw.v, (short)0, acc[nt].v, false, false);
    }
  }

  // h and x-h back into LDS (each wave touches only its own 16 rows)
#pragma unroll
  for (int nt = 0; nt < 8; ++nt) {
    const float bv = bupd[nt * 16 + n];
#pragma unroll
    for (int v = 0; v < 8; ++v) {
      const int rloc = mrow0 + v + 8 * hf;
      float h = acc[nt].f[v] + bv;
      h = h > 0.f ? h : 0.f;
      const float xv = bf2f(sA[rloc * 384 + nt * 16 + n]);
      sA[rloc * 384 + 128 + nt * 16 + n] = f2bf(h);
      sA[rloc * 384 + 256 + nt * 16 + n] = f2bf(xv - h);
    }
  }

  // ---- GEMM2: beta = sigmoid([x|h|x-h] @ Wgate + bgate); blend ----
  Acc8 acc2[8];
#pragma unroll
  for (int nt = 0; nt < 8; ++nt)
#pragma unroll
    for (int j = 0; j < 8; ++j) acc2[nt].f[j] = 0.f;

  for (int kb = 0; kb < 12; ++kb) {
    const u16* ap = sA + rowIdx * 384 + kb * 32 + hf * 8;
    Frag16 af;
    af.q[0] = *(const v4u*)(ap);
    af.q[1] = *(const v4u*)(ap + 16);
#pragma unroll
    for (int nt = 0; nt < 8; ++nt) {
      Frag16 bw;
      const v4u* bp = (const v4u*)wgate + ((size_t)(nt * 12 + kb) * 32 + lane) * 2;
      bw.q[0] = bp[0];
      bw.q[1] = bp[1];
      acc2[nt].v = __builtin_amdgcn_wmma_f32_16x16x32_bf16(
          false, af.v, false, bw.v, (short)0, acc2[nt].v, false, false);
    }
  }

#pragma unroll
  for (int nt = 0; nt < 8; ++nt) {
    const float bv = bgate[nt * 16 + n];
#pragma unroll
    for (int v = 0; v < 8; ++v) {
      const int rloc = mrow0 + v + 8 * hf;
      const int grow = base + rloc;
      if (grow < rows) {
        const float beta = sigmoidf_(acc2[nt].f[v] + bv);
        const float xv = bf2f(sA[rloc * 384 + nt * 16 + n]);
        const float hv = bf2f(sA[rloc * 384 + 128 + nt * 16 + n]);
        outbuf[(size_t)grow * 128 + nt * 16 + n] = f2bf(beta * xv + (1.f - beta) * hv);
      }
    }
  }
}

// ---------------------------------------------------------------------------
// Final fusion: beta = sigmoid([a ‖ b] @ W + bias); out = beta*a + (1-beta)*b
// fp32 output into d_out.
// ---------------------------------------------------------------------------
__global__ __launch_bounds__(128)
void fusion_blend(int rows,
                  const u16* __restrict__ a,
                  const u16* __restrict__ b,
                  const u16* __restrict__ wpk,     // packed KB=8
                  const float* __restrict__ bias,
                  float* __restrict__ out) {
  __shared__ __align__(16) u16 sA[64 * 256];
  const int t = threadIdx.x;
  const int base = blockIdx.x * 64;

  for (int i = t; i < 64 * 128; i += 128) {
    const int r = i >> 7, w = i & 127;
    const int grow = base + r;
    u32 val = 0;
    if (grow < rows) {
      if (w < 64) val = ((const u32*)(a + (size_t)grow * 128))[w];
      else        val = ((const u32*)(b + (size_t)grow * 128))[w - 64];
    }
    ((u32*)sA)[i] = val;
  }
  __syncthreads();

  const int lane  = t & 31;
  const int wave  = t >> 5;
  const int mrow0 = wave * 16;
  const int mA    = lane & 15;
  const int hf    = lane >> 4;
  const u16* aRow = sA + (mrow0 + mA) * 256;
  const int n = lane & 15;

  Acc8 acc[8];
#pragma unroll
  for (int nt = 0; nt < 8; ++nt)
#pragma unroll
    for (int j = 0; j < 8; ++j) acc[nt].f[j] = 0.f;

  for (int kb = 0; kb < 8; ++kb) {
    Frag16 af;
    const u16* ap = aRow + kb * 32 + hf * 8;
    af.q[0] = *(const v4u*)(ap);
    af.q[1] = *(const v4u*)(ap + 16);
#pragma unroll
    for (int nt = 0; nt < 8; ++nt) {
      Frag16 bw;
      const v4u* bp = (const v4u*)wpk + ((size_t)((nt << 3) + kb) * 32 + lane) * 2;
      bw.q[0] = bp[0];
      bw.q[1] = bp[1];
      acc[nt].v = __builtin_amdgcn_wmma_f32_16x16x32_bf16(
          false, af.v, false, bw.v, (short)0, acc[nt].v, false, false);
    }
  }

#pragma unroll
  for (int nt = 0; nt < 8; ++nt) {
    const float bv = bias[nt * 16 + n];
#pragma unroll
    for (int v = 0; v < 8; ++v) {
      const int rloc = mrow0 + v + 8 * hf;
      const int grow = base + rloc;
      if (grow < rows) {
        const float beta = sigmoidf_(acc[nt].f[v] + bv);
        const float xv = bf2f(sA[rloc * 256 + nt * 16 + n]);
        const float fv = bf2f(sA[rloc * 256 + 128 + nt * 16 + n]);
        out[(size_t)grow * 128 + nt * 16 + n] = beta * xv + (1.f - beta) * fv;
      }
    }
  }
}

// ---------------------------------------------------------------------------
extern "C" void kernel_launch(void* const* d_in, const int* in_sizes, int n_in,
                              void* d_out, int out_size, void* d_ws, size_t ws_size,
                              hipStream_t stream) {
  (void)n_in; (void)out_size; (void)ws_size;

  const int N   = in_sizes[0] / 64;
  const int E   = in_sizes[1] / 16;
  const int BE  = in_sizes[2];
  const int FE  = in_sizes[3] / 16;
  const int BFE = in_sizes[4];

  const float* x_in     = (const float*)d_in[0];
  const float* ea_in    = (const float*)d_in[1];
  const float* baa_in   = (const float*)d_in[2];
  const float* fea_in   = (const float*)d_in[3];
  const float* bfa_in   = (const float*)d_in[4];
  const float* lin_a_w  = (const float*)d_in[5];   const float* lin_a_b  = (const float*)d_in[6];
  const float* lin_b_w  = (const float*)d_in[7];   const float* lin_b_b  = (const float*)d_in[8];
  const float* lin_c_w  = (const float*)d_in[9];   const float* lin_c_b  = (const float*)d_in[10];
  const float* lin_gate_w    = (const float*)d_in[11]; const float* lin_gate_b    = (const float*)d_in[12];
  const float* geo_gate_w    = (const float*)d_in[13]; const float* geo_gate_b    = (const float*)d_in[14];
  const float* fusion_gate_w = (const float*)d_in[15]; const float* fusion_gate_b = (const float*)d_in[16];
  const float* gc_msg_w   = (const float*)d_in[17]; const float* gc_msg_b   = (const float*)d_in[18];
  const float* gc_upd_w   = (const float*)d_in[19]; const float* gc_upd_b   = (const float*)d_in[20];
  const float* geoc_msg_w = (const float*)d_in[21]; const float* geoc_msg_b = (const float*)d_in[22];
  const float* geoc_upd_w = (const float*)d_in[23]; const float* geoc_upd_b = (const float*)d_in[24];
  const int* ei       = (const int*)d_in[25];
  const int* ba_ei    = (const int*)d_in[26];
  const int* fra_ei   = (const int*)d_in[27];
  const int* bafra_ei = (const int*)d_in[28];

  // workspace carve-out (regions reused across phases)
  char* ws = (char*)d_ws;
  size_t off = 0;
  auto carve = [&](size_t bytes) -> char* {
    char* p = ws + off;
    off = (off + bytes + 255) & ~(size_t)255;
    return p;
  };
  u16*   XE   = (u16*)carve((size_t)N * 128 * 2);    // embedded nodes (both branches)
  u16*   XOUT = (u16*)carve((size_t)N * 128 * 2);    // branch-1 node output
  u16*   EA   = (u16*)carve((size_t)E * 128 * 2);    // ea   / fea
  u16*   BAA  = (u16*)carve((size_t)BE * 128 * 2);   // baa  / bfa
  u16*   EA2  = (u16*)carve((size_t)E * 128 * 2);    // ea'  / fea' / fx_out
  float* AGGE = (float*)carve((size_t)E * 128 * 4);  // edge-level segment sums
  float* AGGN = (float*)carve((size_t)N * 128 * 4);  // node-level segment sums
  u16* PK_GEOC_MSG = (u16*)carve(256 * 128 * 2);
  u16* PK_GEOC_UPD = (u16*)carve(256 * 128 * 2);
  u16* PK_GEO_GATE = (u16*)carve(384 * 128 * 2);
  u16* PK_GC_MSG   = (u16*)carve(256 * 128 * 2);
  u16* PK_GC_UPD   = (u16*)carve(256 * 128 * 2);
  u16* PK_LIN_GATE = (u16*)carve(384 * 128 * 2);
  u16* PK_FUSION   = (u16*)carve(256 * 128 * 2);

  // --- pack weights into WMMA B-fragment layout (bf16) ---
  pack_w<<<(8 * 8 * 32 + 127) / 128, 128, 0, stream>>>(geoc_msg_w, 8, PK_GEOC_MSG);
  pack_w<<<(8 * 8 * 32 + 127) / 128, 128, 0, stream>>>(geoc_upd_w, 8, PK_GEOC_UPD);
  pack_w<<<(8 * 12 * 32 + 127) / 128, 128, 0, stream>>>(geo_gate_w, 12, PK_GEO_GATE);
  pack_w<<<(8 * 8 * 32 + 127) / 128, 128, 0, stream>>>(gc_msg_w, 8, PK_GC_MSG);
  pack_w<<<(8 * 8 * 32 + 127) / 128, 128, 0, stream>>>(gc_upd_w, 8, PK_GC_UPD);
  pack_w<<<(8 * 12 * 32 + 127) / 128, 128, 0, stream>>>(lin_gate_w, 12, PK_LIN_GATE);
  pack_w<<<(8 * 8 * 32 + 127) / 128, 128, 0, stream>>>(fusion_gate_w, 8, PK_FUSION);

  // --- embedders ---
  embed_relu<<<N, 128, 0, stream>>>(N, 64, x_in, lin_a_w, lin_a_b, XE);
  embed_relu<<<E, 128, 0, stream>>>(E, 16, ea_in, lin_b_w, lin_b_b, EA);
  embed_relu<<<BE, 128, 0, stream>>>(BE, 1, baa_in, lin_c_w, lin_c_b, BAA);

  // ===== branch 1 (base graph) =====
  zero_f32<<<(int)(((size_t)E * 128 + 255) / 256), 256, 0, stream>>>(AGGE, (size_t)E * 128);
  msg_gemm_scatter<<<(BE + 63) / 64, 128, 0, stream>>>(
      BE, EA, BAA, ba_ei, ba_ei + BE, PK_GEOC_MSG, geoc_msg_b, AGGE);
  upd_gate_blend<<<(E + 63) / 64, 128, 0, stream>>>(
      E, EA, AGGE, PK_GEOC_UPD, geoc_upd_b, PK_GEO_GATE, geo_gate_b, EA2);
  zero_f32<<<(int)(((size_t)N * 128 + 255) / 256), 256, 0, stream>>>(AGGN, (size_t)N * 128);
  msg_gemm_scatter<<<(E + 63) / 64, 128, 0, stream>>>(
      E, XE, EA2, ei, ei + E, PK_GC_MSG, gc_msg_b, AGGN);
  upd_gate_blend<<<(N + 63) / 64, 128, 0, stream>>>(
      N, XE, AGGN, PK_GC_UPD, gc_upd_b, PK_LIN_GATE, lin_gate_b, XOUT);

  // ===== branch 2 (fragment graph) =====
  embed_relu<<<FE, 128, 0, stream>>>(FE, 16, fea_in, lin_b_w, lin_b_b, EA);
  embed_relu<<<BFE, 128, 0, stream>>>(BFE, 1, bfa_in, lin_c_w, lin_c_b, BAA);
  zero_f32<<<(int)(((size_t)FE * 128 + 255) / 256), 256, 0, stream>>>(AGGE, (size_t)FE * 128);
  msg_gemm_scatter<<<(BFE + 63) / 64, 128, 0, stream>>>(
      BFE, EA, BAA, bafra_ei, bafra_ei + BFE, PK_GEOC_MSG, geoc_msg_b, AGGE);
  upd_gate_blend<<<(FE + 63) / 64, 128, 0, stream>>>(
      FE, EA, AGGE, PK_GEOC_UPD, geoc_upd_b, PK_GEO_GATE, geo_gate_b, EA2);
  zero_f32<<<(int)(((size_t)N * 128 + 255) / 256), 256, 0, stream>>>(AGGN, (size_t)N * 128);
  msg_gemm_scatter<<<(FE + 63) / 64, 128, 0, stream>>>(
      FE, XE, EA2, fra_ei, fra_ei + FE, PK_GC_MSG, gc_msg_b, AGGN);
  upd_gate_blend<<<(N + 63) / 64, 128, 0, stream>>>(
      N, XE, AGGN, PK_GC_UPD, gc_upd_b, PK_LIN_GATE, lin_gate_b, EA2);  // fx_out -> EA2

  // ===== fusion gate =====
  fusion_blend<<<(N + 63) / 64, 128, 0, stream>>>(
      N, XOUT, EA2, PK_FUSION, fusion_gate_b, (float*)d_out);
}
